// LRUCell_56221121905330
// MI455X (gfx1250) — matrix operand order repaired and checked by reading, not compiled, not run on hardware
//
#include <hip/hip_runtime.h>

#define B_DIM   8192
#define IN_DIM  1024
#define H_DIM   2048
#define OUT_DIM 1024
#define BK      32
#define LP      40   // LDS row pitch (elements): 80B, 16B-aligned, conflict-free

typedef __attribute__((ext_vector_type(16))) __bf16 v16bf;
typedef __attribute__((ext_vector_type(8)))  __bf16 v8bf;
typedef __attribute__((ext_vector_type(8)))  float  v8f;

#if __has_builtin(__builtin_amdgcn_sched_barrier)
#define SFENCE() __builtin_amdgcn_sched_barrier(0)
#else
#define SFENCE()
#endif

// fp32 -> bf16, round-to-nearest-even
static __device__ __forceinline__ __bf16 f32_to_bf16_rne(float f) {
    unsigned u = __builtin_bit_cast(unsigned, f);
    u += 0x7FFFu + ((u >> 16) & 1u);
    unsigned short h = (unsigned short)(u >> 16);
    return __builtin_bit_cast(__bf16, h);
}

// Build a 16-elem fragment from two 16B LDS chunks at row*LP+e0 and +e0+d.
// A-frag (16-bit 16x32): e0=(lane>>4)*8,  d=16  (K quarters {0..7,16..23} / {8..15,24..31})
// B-frag (32x16):        e0=(lane>>4)*8,  d=16  (K quarters {0..7,16..23} / {8..15,24..31})
static __device__ __forceinline__ v16bf lds_frag(const __bf16* s, int row, int e0, int d) {
    const __bf16* q = s + row * LP + e0;
    v8bf lo = *(const v8bf*)(q);
    v8bf hi = *(const v8bf*)(q + d);
    return __builtin_shufflevector(lo, hi, 0,1,2,3,4,5,6,7,8,9,10,11,12,13,14,15);
}

#define WMMA_BF16(a, b, c) \
    __builtin_amdgcn_wmma_f32_16x16x32_bf16(false, (a), false, (b), (short)0, (c), false, false)

// ---------------- prep kernels ----------------

__global__ void cvt_f32_bf16(const float* __restrict__ src, __bf16* __restrict__ dst,
                             int n, float scale) {
    int i = blockIdx.x * blockDim.x + threadIdx.x;
    const int stride = gridDim.x * blockDim.x;
    for (; i < n; i += stride) dst[i] = f32_to_bf16_rne(src[i] * scale);
}

__global__ void compute_w(const float* __restrict__ v_log, const float* __restrict__ t_log,
                          float* __restrict__ wr, float* __restrict__ wi, int n) {
    int i = blockIdx.x * blockDim.x + threadIdx.x;
    if (i < n) {
        float mag = expf(-expf(v_log[i]));
        float ang = expf(t_log[i]);
        wr[i] = mag * cosf(ang);
        wi[i] = mag * sinf(ang);
    }
}

// ---------------- stage 1: hr/hi = x @ B^T + h_prev * w ----------------
// Block 128(M) x 64(N), 8 waves each owning a 32x32 tile (x2 for r/i).
// x / Br / Bi K-slabs double-buffered in LDS; x fragments shared by Br & Bi WMMAs.

__global__ __launch_bounds__(256)
void lru_stage1(const __bf16* __restrict__ xb,
                const __bf16* __restrict__ Brb, const __bf16* __restrict__ Bib,
                const float* __restrict__ h_prev,
                const float* __restrict__ wr, const float* __restrict__ wi,
                __bf16* __restrict__ hrb, __bf16* __restrict__ hib) {
    __shared__ __bf16 sX[2][128 * LP];
    __shared__ __bf16 sR[2][64 * LP];
    __shared__ __bf16 sI[2][64 * LP];

    const int tid  = threadIdx.x;
    const int lane = tid & 31;
    const int wave = tid >> 5;
    const int Mblk = blockIdx.x * 128;
    const int Nblk = blockIdx.y * 64;
    const int wm = (wave & 3) * 32;
    const int wn = (wave >> 2) * 32;

    // staging map: x = 512 16B-chunks (2/thread), weights = 256 chunks (1/thread each)
    const int xc0 = 2 * tid, xc1 = 2 * tid + 1;
    const int xrow0 = xc0 >> 2, xcol0 = (xc0 & 3) * 8;
    const int xrow1 = xc1 >> 2, xcol1 = (xc1 & 3) * 8;
    const int wrow = tid >> 2, wcol = (tid & 3) * 8;

    v8bf gx0, gx1, gr, gi;
    auto load_regs = [&](int k0) {
        gx0 = *(const v8bf*)(xb  + (size_t)(Mblk + xrow0) * IN_DIM + k0 + xcol0);
        gx1 = *(const v8bf*)(xb  + (size_t)(Mblk + xrow1) * IN_DIM + k0 + xcol1);
        gr  = *(const v8bf*)(Brb + (size_t)(Nblk + wrow)  * IN_DIM + k0 + wcol);
        gi  = *(const v8bf*)(Bib + (size_t)(Nblk + wrow)  * IN_DIM + k0 + wcol);
    };
    auto store_lds = [&](int buf) {
        *(v8bf*)(&sX[buf][xrow0 * LP + xcol0]) = gx0;
        *(v8bf*)(&sX[buf][xrow1 * LP + xcol1]) = gx1;
        *(v8bf*)(&sR[buf][wrow * LP + wcol])   = gr;
        *(v8bf*)(&sI[buf][wrow * LP + wcol])   = gi;
    };

    v8f accR[2][2] = {};
    v8f accI[2][2] = {};
    const int la = lane & 15, lh = lane >> 4;

    // grouped: load one B-frag, fire its two WMMAs; one group of lookahead.
    auto compute = [&](int buf) {
        SFENCE();
        v16bf a0 = lds_frag(&sX[buf][0], wm + la,      lh * 8, 16);
        v16bf a1 = lds_frag(&sX[buf][0], wm + 16 + la, lh * 8, 16);
        v16bf r0 = lds_frag(&sR[buf][0], wn + la,      lh * 8, 16);
        v16bf r1 = lds_frag(&sR[buf][0], wn + 16 + la, lh * 8, 16);
        accR[0][0] = WMMA_BF16(a0, r0, accR[0][0]);
        accR[1][0] = WMMA_BF16(a1, r0, accR[1][0]);
        v16bf i0 = lds_frag(&sI[buf][0], wn + la,      lh * 8, 16);
        accR[0][1] = WMMA_BF16(a0, r1, accR[0][1]);
        accR[1][1] = WMMA_BF16(a1, r1, accR[1][1]);
        v16bf i1 = lds_frag(&sI[buf][0], wn + 16 + la, lh * 8, 16);
        accI[0][0] = WMMA_BF16(a0, i0, accI[0][0]);
        accI[1][0] = WMMA_BF16(a1, i0, accI[1][0]);
        accI[0][1] = WMMA_BF16(a0, i1, accI[0][1]);
        accI[1][1] = WMMA_BF16(a1, i1, accI[1][1]);
        SFENCE();
    };

    // software pipeline: global loads for step k+1 overlap WMMAs on step k
    load_regs(0);
    store_lds(0);
    __syncthreads();
    const int NSTEP = IN_DIM / BK;
    int cur = 0;
#pragma unroll 1
    for (int step = 0; step < NSTEP; ++step) {
        if (step + 1 < NSTEP) load_regs((step + 1) * BK);
        compute(cur);
        __syncthreads();
        if (step + 1 < NSTEP) store_lds(cur ^ 1);
        __syncthreads();
        cur ^= 1;
    }

    // epilogue: fuse h_prev * w, emit bf16 hr/hi
    const int nbase = Nblk + wn + la;
    const int mhalf = lh << 3;
#pragma unroll
    for (int mi = 0; mi < 2; ++mi)
#pragma unroll
        for (int ni = 0; ni < 2; ++ni) {
            const int n = nbase + ni * 16;
            const float wrn = wr[n];
            const float win = wi[n];
#pragma unroll
            for (int r = 0; r < 8; ++r) {
                const int m = Mblk + wm + mi * 16 + mhalf + r;
                const size_t idx = (size_t)m * H_DIM + n;
                const float hp = h_prev[idx];
                hrb[idx] = f32_to_bf16_rne(accR[mi][ni][r] + hp * wrn);
                hib[idx] = f32_to_bf16_rne(accI[mi][ni][r] + hp * win);
            }
        }
}

// ---------------- stage 2: out = hr @ Cr^T + hi @ (-Ci)^T ----------------

__global__ __launch_bounds__(256)
void lru_stage2(const __bf16* __restrict__ hrb, const __bf16* __restrict__ hib,
                const __bf16* __restrict__ Crb, const __bf16* __restrict__ Cinb,
                float* __restrict__ out) {
    __shared__ __bf16 sHr[2][128 * LP];
    __shared__ __bf16 sHi[2][128 * LP];
    __shared__ __bf16 sCr[2][64 * LP];
    __shared__ __bf16 sCi[2][64 * LP];

    const int tid  = threadIdx.x;
    const int lane = tid & 31;
    const int wave = tid >> 5;
    const int Mblk = blockIdx.x * 128;
    const int Nblk = blockIdx.y * 64;
    const int wm = (wave & 3) * 32;
    const int wn = (wave >> 2) * 32;

    const int xc0 = 2 * tid, xc1 = 2 * tid + 1;
    const int xrow0 = xc0 >> 2, xcol0 = (xc0 & 3) * 8;
    const int xrow1 = xc1 >> 2, xcol1 = (xc1 & 3) * 8;
    const int wrow = tid >> 2, wcol = (tid & 3) * 8;

    v8bf ghr0, ghr1, ghi0, ghi1, gcr, gci;
    auto load_regs = [&](int k0) {
        ghr0 = *(const v8bf*)(hrb  + (size_t)(Mblk + xrow0) * H_DIM + k0 + xcol0);
        ghr1 = *(const v8bf*)(hrb  + (size_t)(Mblk + xrow1) * H_DIM + k0 + xcol1);
        ghi0 = *(const v8bf*)(hib  + (size_t)(Mblk + xrow0) * H_DIM + k0 + xcol0);
        ghi1 = *(const v8bf*)(hib  + (size_t)(Mblk + xrow1) * H_DIM + k0 + xcol1);
        gcr  = *(const v8bf*)(Crb  + (size_t)(Nblk + wrow)  * H_DIM + k0 + wcol);
        gci  = *(const v8bf*)(Cinb + (size_t)(Nblk + wrow)  * H_DIM + k0 + wcol);
    };
    auto store_lds = [&](int buf) {
        *(v8bf*)(&sHr[buf][xrow0 * LP + xcol0]) = ghr0;
        *(v8bf*)(&sHr[buf][xrow1 * LP + xcol1]) = ghr1;
        *(v8bf*)(&sHi[buf][xrow0 * LP + xcol0]) = ghi0;
        *(v8bf*)(&sHi[buf][xrow1 * LP + xcol1]) = ghi1;
        *(v8bf*)(&sCr[buf][wrow * LP + wcol])   = gcr;
        *(v8bf*)(&sCi[buf][wrow * LP + wcol])   = gci;
    };

    v8f acc[2][2] = {};
    const int la = lane & 15, lh = lane >> 4;

    auto compute = [&](int buf) {
        SFENCE();
        v16bf ar0 = lds_frag(&sHr[buf][0], wm + la,      lh * 8, 16);
        v16bf ar1 = lds_frag(&sHr[buf][0], wm + 16 + la, lh * 8, 16);
        v16bf br0 = lds_frag(&sCr[buf][0], wn + la,      lh * 8, 16);
        v16bf br1 = lds_frag(&sCr[buf][0], wn + 16 + la, lh * 8, 16);
        acc[0][0] = WMMA_BF16(ar0, br0, acc[0][0]);
        acc[1][0] = WMMA_BF16(ar1, br0, acc[1][0]);
        v16bf ai0 = lds_frag(&sHi[buf][0], wm + la,      lh * 8, 16);
        v16bf ai1 = lds_frag(&sHi[buf][0], wm + 16 + la, lh * 8, 16);
        acc[0][1] = WMMA_BF16(ar0, br1, acc[0][1]);
        acc[1][1] = WMMA_BF16(ar1, br1, acc[1][1]);
        v16bf bi0 = lds_frag(&sCi[buf][0], wn + la,      lh * 8, 16);
        v16bf bi1 = lds_frag(&sCi[buf][0], wn + 16 + la, lh * 8, 16);
        acc[0][0] = WMMA_BF16(ai0, bi0, acc[0][0]);
        acc[1][0] = WMMA_BF16(ai1, bi0, acc[1][0]);
        acc[0][1] = WMMA_BF16(ai0, bi1, acc[0][1]);
        acc[1][1] = WMMA_BF16(ai1, bi1, acc[1][1]);
        SFENCE();
    };

    load_regs(0);
    store_lds(0);
    __syncthreads();
    const int NSTEP = H_DIM / BK;
    int cur = 0;
#pragma unroll 1
    for (int step = 0; step < NSTEP; ++step) {
        if (step + 1 < NSTEP) load_regs((step + 1) * BK);
        compute(cur);
        __syncthreads();
        if (step + 1 < NSTEP) store_lds(cur ^ 1);
        __syncthreads();
        cur ^= 1;
    }

    const int nbase = Nblk + wn + la;
    const int mhalf = lh << 3;
#pragma unroll
    for (int mi = 0; mi < 2; ++mi)
#pragma unroll
        for (int ni = 0; ni < 2; ++ni)
#pragma unroll
            for (int r = 0; r < 8; ++r) {
                const int m = Mblk + wm + mi * 16 + mhalf + r;
                out[(size_t)m * OUT_DIM + nbase + ni * 16] = acc[mi][ni][r];
            }
}

// ---------------- launcher ----------------

extern "C" void kernel_launch(void* const* d_in, const int* in_sizes, int n_in,
                              void* d_out, int out_size, void* d_ws, size_t ws_size,
                              hipStream_t stream) {
    const float* x      = (const float*)d_in[0];
    const float* h_prev = (const float*)d_in[1];
    const float* Br     = (const float*)d_in[2];
    const float* Bi     = (const float*)d_in[3];
    const float* Cr     = (const float*)d_in[4];
    const float* Ci     = (const float*)d_in[5];
    const float* v_log  = (const float*)d_in[6];
    const float* t_log  = (const float*)d_in[7];
    float* out = (float*)d_out;

    // carve workspace (all sub-buffers 256B aligned) — total ~101 MB
    char* w = (char*)d_ws;
    size_t off = 0;
    auto take = [&](size_t bytes) -> char* {
        char* p = w + off;
        off += (bytes + 255) & ~(size_t)255;
        return p;
    };
    __bf16* xb   = (__bf16*)take((size_t)B_DIM * IN_DIM * 2);
    __bf16* Brb  = (__bf16*)take((size_t)H_DIM * IN_DIM * 2);
    __bf16* Bib  = (__bf16*)take((size_t)H_DIM * IN_DIM * 2);
    __bf16* Crb  = (__bf16*)take((size_t)OUT_DIM * H_DIM * 2);
    __bf16* Cinb = (__bf16*)take((size_t)OUT_DIM * H_DIM * 2);  // holds -Ci
    __bf16* hrb  = (__bf16*)take((size_t)B_DIM * H_DIM * 2);
    __bf16* hib  = (__bf16*)take((size_t)B_DIM * H_DIM * 2);
    float*  wr   = (float*)take((size_t)H_DIM * 4);
    float*  wi   = (float*)take((size_t)H_DIM * 4);

    cvt_f32_bf16<<<4096, 256, 0, stream>>>(x,  xb,  B_DIM * IN_DIM,   1.0f);
    cvt_f32_bf16<<<2048, 256, 0, stream>>>(Br, Brb, H_DIM * IN_DIM,   1.0f);
    cvt_f32_bf16<<<2048, 256, 0, stream>>>(Bi, Bib, H_DIM * IN_DIM,   1.0f);
    cvt_f32_bf16<<<2048, 256, 0, stream>>>(Cr, Crb, OUT_DIM * H_DIM,  1.0f);
    cvt_f32_bf16<<<2048, 256, 0, stream>>>(Ci, Cinb, OUT_DIM * H_DIM, -1.0f);
    compute_w<<<(H_DIM + 255) / 256, 256, 0, stream>>>(v_log, t_log, wr, wi, H_DIM);

    lru_stage1<<<dim3(B_DIM / 128, H_DIM / 64), 256, 0, stream>>>(
        xb, Brb, Bib, h_prev, wr, wi, hrb, hib);
    lru_stage2<<<dim3(B_DIM / 128, OUT_DIM / 64), 256, 0, stream>>>(
        hrb, hib, Crb, Cinb, out);
}
